// LSTMLightweight_39779987096022
// MI455X (gfx1250) — matrix operand order repaired and checked by reading, not compiled, not run on hardware
//
#include <hip/hip_runtime.h>
#include <hip/hip_bf16.h>
#include <stdint.h>

// ---------------------------------------------------------------------------
// Fused 2-layer LSTM (H=16) over T=256, batch 8192, + linear_in and FC head.
// One wave32 owns 16 batch elements. Per timestep per layer:
//   gates^T(64x16) = W(64x32) x [x_feat ; h_prev]^T(32x16)   via 4x
//   v_wmma_f32_16x16x32_f16  (bias folded into the C operand).
// h feedback stays in registers; layout fixup = ds_swizzle XOR16 + cvt_pkrtz.
// Activations use v_tanh_f32 when the builtin exists (TRANS op, co-executes
// with VALU/WMMA), else exp/rcp fallback.
// ---------------------------------------------------------------------------

typedef __attribute__((ext_vector_type(16))) _Float16 v16h;
typedef __attribute__((ext_vector_type(8)))  float    v8f;
typedef __attribute__((ext_vector_type(2)))  __fp16   fp16x2;

#define BATCH 8192
#define TSEQ  256
#define HID   16

union PK { fp16x2 h; unsigned u; };
union AB { v16h v; _Float16 h[16]; unsigned u[8]; };

__device__ __forceinline__ unsigned pk(float a, float b) {
  PK p; p.h = __builtin_amdgcn_cvt_pkrtz(a, b); return p.u;
}
// swap lane halves (lane ^ 16) within the wave
__device__ __forceinline__ unsigned swz16(unsigned x) {
  return (unsigned)__builtin_amdgcn_ds_swizzle((int)x, 0x401f);
}

#if __has_builtin(__builtin_amdgcn_tanhf)
  #define HAVE_TANH 1
  __device__ __forceinline__ float tanh_f(float x) { return __builtin_amdgcn_tanhf(x); }
#elif __has_builtin(__builtin_amdgcn_tanh_f32)
  #define HAVE_TANH 1
  __device__ __forceinline__ float tanh_f(float x) { return __builtin_amdgcn_tanh_f32(x); }
#else
  #define HAVE_TANH 0
  __device__ __forceinline__ float tanh_f(float x) {
    float e = __expf(-2.0f * x);
    return (1.0f - e) * __builtin_amdgcn_rcpf(1.0f + e);
  }
#endif

__device__ __forceinline__ float sigm(float x) {
#if HAVE_TANH
  return fmaf(0.5f, tanh_f(0.5f * x), 0.5f);     // mul + v_tanh + fma
#else
  return __builtin_amdgcn_rcpf(1.0f + __expf(-x));
#endif
}

__device__ __forceinline__ v8f wmma(v16h a, v16h b, v8f c) {
  return __builtin_amdgcn_wmma_f32_16x16x32_f16(
      /*neg_a=*/false, a, /*neg_b=*/false, b,
      /*c_mod=*/(short)0, c, /*reuse_a=*/false, /*reuse_b=*/false);
}

__global__ __launch_bounds__(128) void lstm_fused(
    const float* __restrict__ x,
    const float* __restrict__ lin_w, const float* __restrict__ lin_b,
    const float* __restrict__ wih0,  const float* __restrict__ whh0,
    const float* __restrict__ bih0,  const float* __restrict__ bhh0,
    const float* __restrict__ wih1,  const float* __restrict__ whh1,
    const float* __restrict__ bih1,  const float* __restrict__ bhh1,
    const float* __restrict__ fchw,  const float* __restrict__ fchb,
    const float* __restrict__ fcow,  const float* __restrict__ fcob,
    float* __restrict__ out)
{
  __shared__ float xT[4][TSEQ * HID];   // 64 KB: per-wave 16x256 x slab
  const int  tid  = threadIdx.x;
  const int  wv   = tid >> 5;
  const int  lane = tid & 31;
  const int  lo   = lane & 15;
  const bool hi   = lane >= 16;
  const int  bb   = (blockIdx.x * 4 + wv) * 16;   // batch base of this wave
  float* xs = xT[wv];

  // Stage x[bb..bb+15][0..255] into LDS, coalesced over t.
  for (int idx = lane; idx < HID * TSEQ; idx += 32) {
    int i = idx >> 8, t = idx & 255;
    xs[t * HID + i] = x[(bb + i) * TSEQ + t];
  }

  const float* wihL[2] = { wih0, wih1 };
  const float* whhL[2] = { whh0, whh1 };
  const float* bihL[2] = { bih0, bih1 };
  const float* bhhL[2] = { bhh0, bhh1 };

  // A operands: 4 gate tiles (16 gates x 32 K) per layer, f16.
  // A layout: lane(0-15)=row M; VGPR0-7 K=0..15 (lanes<16) / 16..31 (lanes>=16).
  //           K<16 -> w_ih, K>=16 -> w_hh.
  AB  A[2][4];
  v8f bias[2][4];
  #pragma unroll
  for (int l = 0; l < 2; ++l) {
    #pragma unroll
    for (int j = 0; j < 4; ++j) {
      const int row = j * 16 + lo;
      const float* src = hi ? whhL[l] : wihL[l];
      #pragma unroll
      for (int v = 0; v < 4; ++v) {
        A[l][j].h[2*v]     = (_Float16)src[row * 16 + 2*v];
        A[l][j].h[2*v+1]   = (_Float16)src[row * 16 + 2*v + 1];
        A[l][j].h[8+2*v]   = (_Float16)src[row * 16 + 8 + 2*v];
        A[l][j].h[8+2*v+1] = (_Float16)src[row * 16 + 8 + 2*v + 1];
      }
      #pragma unroll
      for (int r = 0; r < 8; ++r) {       // bias in C/D layout (row-indexed)
        int g = j * 16 + r + (hi ? 8 : 0);
        bias[l][j][r] = bihL[l][g] + bhhL[l][g];
      }
    }
  }

  float lw[16], lb[16];
  #pragma unroll
  for (int k = 0; k < 16; ++k) { lw[k] = lin_w[k]; lb[k] = lin_b[k]; }

  // D-layout state: VGPR r = hidden r (lanes<16) / r+8 (lanes>=16), batch=lane.
  v8f c0 = {}, h0v = {}, c1 = {}, h1v = {};

  for (int t = 0; t < TSEQ; ++t) {
    const float xv = xs[t * HID + lo];

    // ---------------- layer 0 ----------------
    // B operand: lanes<16 hold K=0..15 (linear_in features), lanes>=16 hold
    // K=16..31 (h0 rows), batch column = lane&15.
    AB b0;
    #pragma unroll
    for (int v = 0; v < 4; ++v) {
      unsigned hp = swz16(pk(h0v[2*v], h0v[2*v+1]));      // rows 2v,2v+1 -> hi half
      float f0 = fmaxf(fmaf(xv, lw[2*v],   lb[2*v]),   0.0f);
      float f1 = fmaxf(fmaf(xv, lw[2*v+1], lb[2*v+1]), 0.0f);
      b0.u[v] = hi ? hp : pk(f0, f1);
    }
    #pragma unroll
    for (int v = 4; v < 8; ++v) {
      unsigned hp = pk(h0v[2*v-8], h0v[2*v-7]);           // rows 2v,2v+1 already hi
      float f0 = fmaxf(fmaf(xv, lw[2*v],   lb[2*v]),   0.0f);
      float f1 = fmaxf(fmaf(xv, lw[2*v+1], lb[2*v+1]), 0.0f);
      b0.u[v] = hi ? hp : pk(f0, f1);
    }
    v8f gi = wmma(A[0][0].v, b0.v, bias[0][0]);
    v8f gf = wmma(A[0][1].v, b0.v, bias[0][1]);
    v8f gg = wmma(A[0][2].v, b0.v, bias[0][2]);
    v8f go = wmma(A[0][3].v, b0.v, bias[0][3]);
    #pragma unroll
    for (int r = 0; r < 8; ++r) {
      float cc = sigm(gf[r]) * c0[r] + sigm(gi[r]) * tanh_f(gg[r]);
      c0[r]  = cc;
      h0v[r] = sigm(go[r]) * tanh_f(cc);
    }

    // ---------------- layer 1 ----------------
    // B operand: lanes<16 hold K=0..15 = h0 (current), lanes>=16 K=16..31 = h1.
    AB b1;
    #pragma unroll
    for (int v = 0; v < 4; ++v) {
      unsigned lp = pk(h0v[2*v], h0v[2*v+1]);             // rows 2v,2v+1, lo half
      unsigned hq = swz16(pk(h1v[2*v], h1v[2*v+1]));      // -> hi half
      b1.u[v] = hi ? hq : lp;
    }
    #pragma unroll
    for (int v = 4; v < 8; ++v) {
      unsigned lp = swz16(pk(h0v[2*v-8], h0v[2*v-7]));    // rows 2v.. from hi -> lo
      unsigned hq = pk(h1v[2*v-8], h1v[2*v-7]);           // already hi half
      b1.u[v] = hi ? hq : lp;
    }
    gi = wmma(A[1][0].v, b1.v, bias[1][0]);
    gf = wmma(A[1][1].v, b1.v, bias[1][1]);
    gg = wmma(A[1][2].v, b1.v, bias[1][2]);
    go = wmma(A[1][3].v, b1.v, bias[1][3]);
    #pragma unroll
    for (int r = 0; r < 8; ++r) {
      float cc = sigm(gf[r]) * c1[r] + sigm(gi[r]) * tanh_f(gg[r]);
      c1[r]  = cc;
      h1v[r] = sigm(go[r]) * tanh_f(cc);
    }
  }

  // -------- FC head: out = fc_o( relu(fc_h(h1)) ) , per batch column --------
  #pragma unroll
  for (int r = 0; r < 8; ++r) {                 // dump h1 to LDS, stride 17
    int row = r + (hi ? 8 : 0);
    xs[row * 17 + lo] = h1v[r];
  }
  if (lane < 16) {
    float acc = fcob[0];
    #pragma unroll
    for (int j = 0; j < 8; ++j) {
      float s = fchb[j];
      #pragma unroll
      for (int k = 0; k < 16; ++k)
        s = fmaf(fchw[j * 16 + k], xs[k * 17 + lo], s);
      acc = fmaf(fcow[j], fmaxf(s, 0.0f), acc);
    }
    out[bb + lo] = acc;
  }
}

extern "C" void kernel_launch(void* const* d_in, const int* in_sizes, int n_in,
                              void* d_out, int out_size, void* d_ws, size_t ws_size,
                              hipStream_t stream) {
  const float* x    = (const float*)d_in[0];
  const float* linw = (const float*)d_in[1];
  const float* linb = (const float*)d_in[2];
  const float* wih0 = (const float*)d_in[3];
  const float* whh0 = (const float*)d_in[4];
  const float* bih0 = (const float*)d_in[5];
  const float* bhh0 = (const float*)d_in[6];
  const float* wih1 = (const float*)d_in[7];
  const float* whh1 = (const float*)d_in[8];
  const float* bih1 = (const float*)d_in[9];
  const float* bhh1 = (const float*)d_in[10];
  const float* fchw = (const float*)d_in[11];
  const float* fchb = (const float*)d_in[12];
  const float* fcow = (const float*)d_in[13];
  const float* fcob = (const float*)d_in[14];
  float* out = (float*)d_out;

  dim3 grid(BATCH / 64);   // 4 waves/block * 16 batch/wave
  dim3 block(128);
  hipLaunchKernelGGL(lstm_fused, grid, block, 0, stream,
                     x, linw, linb, wih0, whh0, bih0, bhh0,
                     wih1, whh1, bih1, bhh1, fchw, fchb, fcow, fcob, out);
}
